// GATModel_23880018165825
// MI455X (gfx1250) — matrix operand order repaired and checked by reading, not compiled, not run on hardware
//
#include <hip/hip_runtime.h>
#include <hip/hip_bf16.h>
#include <stdint.h>

// ---------------- problem constants (match reference) ----------------
#define N_NODES   100000                // divisible by 16 -> no row masking in GEMM
#define N_EDGES   1600000
#define E_TOT     (N_EDGES + N_NODES)   // with self loops
#define HID       100
#define KPAD      128                   // K padded for 16x16x32 WMMA
#define NPAD      112                   // 7 col tiles of 16; hx stored with this stride
#define LDSP      144                   // LDS row stride (halfs): 128 + 16 pad (bank spread)
#define LAYERS    6
#define OUTF      64
#define NGRAPHS   512
#define LN_EPS    1e-5f

#define USE_ASYNC_LDS 1                 // gfx1250 global_load_async_to_lds_b128 staging

typedef __attribute__((ext_vector_type(16))) _Float16 v16h;
typedef __attribute__((ext_vector_type(8)))  _Float16 v8h;
typedef __attribute__((ext_vector_type(8)))  float    v8f;

// ---------------- helpers ----------------
__device__ __forceinline__ float wave_sum32(float v) {
#pragma unroll
  for (int off = 16; off > 0; off >>= 1) v += __shfl_xor(v, off, 32);
  return v;
}

// order-preserving float<->uint encoding for atomic max
__device__ __forceinline__ unsigned enc_f32(float f) {
  unsigned u = __float_as_uint(f);
  return (u & 0x80000000u) ? ~u : (u | 0x80000000u);
}
__device__ __forceinline__ float dec_f32(unsigned u) {
  return (u & 0x80000000u) ? __uint_as_float(u & 0x7fffffffu)
                           : __uint_as_float(~u);
}

// ---------------- generic zero fill ----------------
__global__ void fill_zero_kernel(uint32_t* __restrict__ p, size_t n) {
  size_t i = (size_t)blockIdx.x * blockDim.x + threadIdx.x;
  if (i < n) p[i] = 0u;
}

// ---------------- weight prep: Wh[l][n][k] = f16(Ws[l][n][k]), padded ----------------
__global__ void prep_wh_kernel(const float* __restrict__ Ws, _Float16* __restrict__ Wh) {
  int idx = blockIdx.x * blockDim.x + threadIdx.x;
  const int tot = LAYERS * NPAD * KPAD;
  if (idx >= tot) return;
  int k = idx % KPAD;
  int n = (idx / KPAD) % NPAD;
  int l = idx / (KPAD * NPAD);
  float v = (n < HID && k < HID) ? Ws[(size_t)l * HID * HID + (size_t)n * HID + k] : 0.0f;
  Wh[idx] = (_Float16)v;
}

// ---------------- Q[l][k] = sum_j We[l][j][k] * a[l][HID+j]  (ee collapse) ----------------
__global__ void prep_q_kernel(const float* __restrict__ We, const float* __restrict__ a,
                              float* __restrict__ Q) {
  int idx = blockIdx.x * blockDim.x + threadIdx.x;
  if (idx >= LAYERS * KPAD) return;
  int k = idx % KPAD, l = idx / KPAD;
  float acc = 0.0f;
  if (k < HID) {
    const float* Wl = We + (size_t)l * HID * HID;
    const float* a2 = a + (size_t)l * 2 * HID + HID;
#pragma unroll 4
    for (int j = 0; j < HID; ++j) acc += Wl[(size_t)j * HID + k] * a2[j];
  }
  Q[idx] = acc;
}

// ---------------- one-pass edge term: et[l][e] = ea_e . Q_l  (wave per edge) ----------------
__global__ void edgeterm_kernel(const float* __restrict__ ea, const float* __restrict__ Q,
                                float* __restrict__ et) {
  __shared__ float qs[LAYERS * KPAD];
  for (int i = threadIdx.x; i < LAYERS * KPAD; i += blockDim.x) qs[i] = Q[i];
  __syncthreads();
  int wavesPerBlk = blockDim.x >> 5;
  int e    = blockIdx.x * wavesPerBlk + (threadIdx.x >> 5);
  int lane = threadIdx.x & 31;
  if (e >= N_EDGES) return;
  const float* row = ea + (size_t)e * HID;
  if (e + 64 < N_EDGES)                        // gfx1250 global_prefetch_b8
    __builtin_prefetch(ea + (size_t)(e + 64) * HID, 0, 0);
  float acc[LAYERS];
#pragma unroll
  for (int l = 0; l < LAYERS; ++l) acc[l] = 0.0f;
#pragma unroll
  for (int j = 0; j < 4; ++j) {
    int k = lane + 32 * j;
    if (k < HID) {
      float x = row[k];
#pragma unroll
      for (int l = 0; l < LAYERS; ++l) acc[l] += x * qs[l * KPAD + k];
    }
  }
#pragma unroll
  for (int l = 0; l < LAYERS; ++l) {
    float s = wave_sum32(acc[l]);
    if (lane == 0) et[(size_t)l * N_EDGES + e] = s;
  }
}

// ---------------- embedding gather: h, h_half(padded) (wave per node) ----------------
__global__ void embed_kernel(const int* __restrict__ x, const float* __restrict__ emb,
                             float* __restrict__ h, _Float16* __restrict__ hh) {
  int i    = blockIdx.x * (blockDim.x >> 5) + (threadIdx.x >> 5);
  int lane = threadIdx.x & 31;
  if (i >= N_NODES) return;
  int xi = x[i];
  const float* er = emb + (size_t)xi * HID;
#pragma unroll
  for (int j = 0; j < 4; ++j) {
    int k = lane + 32 * j;
    float v = (k < HID) ? er[k] : 0.0f;
    if (k < HID) h[(size_t)i * HID + k] = v;
    hh[(size_t)i * KPAD + k] = (_Float16)v;     // zero padding for k >= HID
  }
}

// ---------------- WMMA GEMM: Y[N,NPAD] = Xh[N,128] @ Whᵀ  (wave per 16-row tile) ----------------
// B panel (28KB) staged in LDS (async-to-LDS copy), bank-padded row stride LDSP.
__global__ __launch_bounds__(128)
void gemm_hx_kernel(const _Float16* __restrict__ Xh, const _Float16* __restrict__ Wh,
                    float* __restrict__ Y, int nrows) {
  __shared__ alignas(32) _Float16 wlds[NPAD * LDSP];

  // cooperative copy of the 112x128 f16 weight panel into LDS
#if USE_ASYNC_LDS
  for (int c = threadIdx.x; c < NPAD * 16; c += blockDim.x) {   // 16B chunks
    int row = c >> 4, seg = c & 15;
    unsigned ldsOff = (unsigned)(uintptr_t)(wlds + (size_t)row * LDSP + seg * 8);
    const _Float16* g = Wh + (size_t)row * KPAD + seg * 8;
    asm volatile("global_load_async_to_lds_b128 %0, %1, off"
                 :: "v"(ldsOff), "v"(g) : "memory");
  }
  asm volatile("s_wait_asynccnt 0" ::: "memory");
#else
  for (int c = threadIdx.x; c < NPAD * 16; c += blockDim.x) {
    int row = c >> 4, seg = c & 15;
    *(uint4*)(wlds + (size_t)row * LDSP + seg * 8) =
        *(const uint4*)(Wh + (size_t)row * KPAD + seg * 8);
  }
#endif
  __syncthreads();

  int rowTile = blockIdx.x * (blockDim.x >> 5) + (threadIdx.x >> 5);
  int lane    = threadIdx.x & 31;
  int rowBase = rowTile * 16;
  if (rowBase >= nrows) return;                 // wave-uniform exit: EXEC stays all-1s

  const _Float16* xrow = Xh + (size_t)(rowBase + (lane & 15)) * KPAD;
  int hi = lane >> 4;                           // half-wave selector

  // A fragments (16x32 f16): lanes 0-15 hold K {0..7,16..23}, lanes 16-31 K {8..15,24..31}
  v16h afrag[4];
#pragma unroll
  for (int kt = 0; kt < 4; ++kt) {
    int kb = kt * 32 + hi * 8;
    v8h lo = *(const v8h*)(xrow + kb);
    v8h hg = *(const v8h*)(xrow + kb + 16);
    afrag[kt] = __builtin_shufflevector(lo, hg, 0,1,2,3,4,5,6,7,8,9,10,11,12,13,14,15);
  }

  int col = (lane & 15);
#pragma unroll
  for (int ct = 0; ct < 7; ++ct) {
    const _Float16* wrow = wlds + (size_t)(ct * 16 + col) * LDSP;
    v8f c = {};
#pragma unroll
    for (int kt = 0; kt < 4; ++kt) {
      // B fragment (32x16 f16): lanes 0-15 K {0..7,16..23}, lanes 16-31 K {8..15,24..31}, col = lane&15
      v8h blo = *(const v8h*)(wrow + kt * 32 + hi * 8);
      v8h bhg = *(const v8h*)(wrow + kt * 32 + hi * 8 + 16);
      v16h b = __builtin_shufflevector(blo, bhg, 0,1,2,3,4,5,6,7,8,9,10,11,12,13,14,15);
      c = __builtin_amdgcn_wmma_f32_16x16x32_f16(false, afrag[kt], false, b,
                                                 (short)0, c, false, false);
    }
    // D: VGPR v -> row v (+8 for upper half-wave); Y padded to NPAD -> unconditional stores
    float* yrow = Y + (size_t)(rowBase + hi * 8) * NPAD + (ct * 16 + col);
#pragma unroll
    for (int v = 0; v < 8; ++v) yrow[(size_t)v * NPAD] = c[v];
  }
}

// ---------------- per-node attention scalars p1 = hx.a1, p2 = hx.a2 (wave per node) ----------------
__global__ void pscal_kernel(const float* __restrict__ hx, const float* __restrict__ a_l,
                             float* __restrict__ p1, float* __restrict__ p2) {
  int i    = blockIdx.x * (blockDim.x >> 5) + (threadIdx.x >> 5);
  int lane = threadIdx.x & 31;
  if (i >= N_NODES) return;
  const float* r = hx + (size_t)i * NPAD;
  float s1 = 0.0f, s2 = 0.0f;
#pragma unroll
  for (int j = 0; j < 4; ++j) {
    int k = lane + 32 * j;
    if (k < HID) {
      float v = r[k];
      s1 += v * a_l[k];
      s2 += v * a_l[HID + k];
    }
  }
  s1 = wave_sum32(s1);
  s2 = wave_sum32(s2);
  if (lane == 0) { p1[i] = s1; p2[i] = s2; }
}

// ---------------- logits + leaky-relu + segment-max (thread per edge) ----------------
__global__ void logits_kernel(const int* __restrict__ ei, const float* __restrict__ et_l,
                              const float* __restrict__ p1, const float* __restrict__ p2,
                              float* __restrict__ z, unsigned* __restrict__ mu) {
  int e = blockIdx.x * blockDim.x + threadIdx.x;
  if (e >= E_TOT) return;
  int src, dst; float ev;
  if (e < N_EDGES) { src = ei[e]; dst = ei[N_EDGES + e]; ev = et_l[e]; }
  else             { src = dst = e - N_EDGES;            ev = 0.0f;   }
  float zv = p1[dst] + p2[src] + ev;
  zv = zv > 0.0f ? zv : 0.2f * zv;              // leaky_relu(0.2)
  z[e] = zv;
  atomicMax(mu + dst, enc_f32(zv));
}

// ---------------- exp + segment-sum (thread per edge) ----------------
__global__ void expsum_kernel(const int* __restrict__ ei, const float* __restrict__ z,
                              const unsigned* __restrict__ mu,
                              float* __restrict__ ex, float* __restrict__ s) {
  int e = blockIdx.x * blockDim.x + threadIdx.x;
  if (e >= E_TOT) return;
  int dst = (e < N_EDGES) ? ei[N_EDGES + e] : (e - N_EDGES);
  float v = __expf(z[e] - dec_f32(mu[dst]));
  ex[e] = v;
  atomicAdd(s + dst, v);
}

// ---------------- out[dst] += hx[src] * alpha (wave per edge; L2-resident atomics) ----------------
__global__ void scatter_kernel(const int* __restrict__ ei, const float* __restrict__ hx,
                               const float* __restrict__ ex, const float* __restrict__ s,
                               float* __restrict__ out) {
  int e    = blockIdx.x * (blockDim.x >> 5) + (threadIdx.x >> 5);
  int lane = threadIdx.x & 31;
  if (e >= E_TOT) return;
  int src, dst;
  if (e < N_EDGES) { src = ei[e]; dst = ei[N_EDGES + e]; }
  else             { src = dst = e - N_EDGES; }
  float alpha = ex[e] / (s[dst] + 1e-16f);
  const float* xr = hx + (size_t)src * NPAD;
  float* orow = out + (size_t)dst * HID;
#pragma unroll
  for (int j = 0; j < 4; ++j) {
    int k = lane + 32 * j;
    if (k < HID) atomicAdd(orow + k, xr[k] * alpha);
  }
}

// ---------------- layernorm + elu + residual; emit f32 + padded f16 (wave per node) ----------------
__global__ void ln_kernel(const float* __restrict__ out,
                          const float* __restrict__ g, const float* __restrict__ b,
                          float* __restrict__ h, _Float16* __restrict__ hh) {
  int i    = blockIdx.x * (blockDim.x >> 5) + (threadIdx.x >> 5);
  int lane = threadIdx.x & 31;
  if (i >= N_NODES) return;
  const float* r = out + (size_t)i * HID;
  float v[4]; float sum = 0.0f;
#pragma unroll
  for (int j = 0; j < 4; ++j) {
    int k = lane + 32 * j;
    v[j] = (k < HID) ? r[k] : 0.0f;
    sum += v[j];
  }
  float mu = wave_sum32(sum) * (1.0f / HID);
  float vs = 0.0f;
#pragma unroll
  for (int j = 0; j < 4; ++j) {
    int k = lane + 32 * j;
    if (k < HID) { float d = v[j] - mu; vs += d * d; }
  }
  float var = wave_sum32(vs) * (1.0f / HID);
  float rs  = rsqrtf(var + LN_EPS);
#pragma unroll
  for (int j = 0; j < 4; ++j) {
    int k = lane + 32 * j;
    float hn = 0.0f;
    if (k < HID) {
      float y = (v[j] - mu) * rs * g[k] + b[k];
      y  = (y > 0.0f) ? y : (__expf(y) - 1.0f);         // elu
      hn = y + h[(size_t)i * HID + k];                  // residual
      h[(size_t)i * HID + k] = hn;
    }
    hh[(size_t)i * KPAD + k] = (_Float16)hn;            // padded f16 for next WMMA
  }
}

// ---------------- mean-pool accumulate (wave per node) ----------------
__global__ void pool_kernel(const int* __restrict__ batch, const float* __restrict__ h,
                            float* __restrict__ pooled, float* __restrict__ cnt) {
  int i    = blockIdx.x * (blockDim.x >> 5) + (threadIdx.x >> 5);
  int lane = threadIdx.x & 31;
  if (i >= N_NODES) return;
  int bg = batch[i];
  const float* r = h + (size_t)i * HID;
  float* pr = pooled + (size_t)bg * HID;
#pragma unroll
  for (int j = 0; j < 4; ++j) {
    int k = lane + 32 * j;
    if (k < HID) atomicAdd(pr + k, r[k]);
  }
  if (lane == 0) atomicAdd(cnt + bg, 1.0f);
}

// ---------------- final FC (thread per (graph, out-feature)) ----------------
__global__ void fc_kernel(const float* __restrict__ pooled, const float* __restrict__ cnt,
                          const float* __restrict__ fw, const float* __restrict__ fb,
                          float* __restrict__ outp) {
  int idx = blockIdx.x * blockDim.x + threadIdx.x;
  if (idx >= NGRAPHS * OUTF) return;
  int o = idx % OUTF, gI = idx / OUTF;
  float c = cnt[gI];
  float inv = 1.0f / fmaxf(c, 1.0f);
  const float* pr = pooled + (size_t)gI * HID;
  const float* wr = fw + (size_t)o * HID;
  float acc = fb[o];
#pragma unroll 4
  for (int k = 0; k < HID; ++k) acc += pr[k] * inv * wr[k];
  outp[(size_t)gI * OUTF + o] = acc;
}

// ---------------- host orchestration ----------------
static inline size_t alignup(size_t x) { return (x + 255) & ~(size_t)255; }

extern "C" void kernel_launch(void* const* d_in, const int* /*in_sizes*/, int /*n_in*/,
                              void* d_out, int /*out_size*/, void* d_ws, size_t /*ws_size*/,
                              hipStream_t stream) {
  const int*   x    = (const int*)  d_in[0];
  const int*   ei   = (const int*)  d_in[1];   // [2][N_EDGES]
  const float* ea   = (const float*)d_in[2];
  const int*   bat  = (const int*)  d_in[3];
  const float* emb  = (const float*)d_in[4];
  const float* Ws   = (const float*)d_in[5];
  const float* We   = (const float*)d_in[6];
  const float* av   = (const float*)d_in[7];
  const float* lng  = (const float*)d_in[8];
  const float* lnb  = (const float*)d_in[9];
  const float* fw   = (const float*)d_in[10];
  const float* fb   = (const float*)d_in[11];
  float* outp = (float*)d_out;

  char* base = (char*)d_ws;
  size_t off = 0;
  auto carve = [&](size_t bytes) { char* p = base + off; off = alignup(off + bytes); return (void*)p; };

  float*    h      = (float*)   carve((size_t)N_NODES * HID * 4);
  float*    hx     = (float*)   carve((size_t)N_NODES * NPAD * 4);   // padded stride
  float*    outb   = (float*)   carve((size_t)N_NODES * HID * 4);
  _Float16* hh     = (_Float16*)carve((size_t)N_NODES * KPAD * 2);
  _Float16* Wh     = (_Float16*)carve((size_t)LAYERS * NPAD * KPAD * 2);
  float*    Q      = (float*)   carve((size_t)LAYERS * KPAD * 4);
  float*    et     = (float*)   carve((size_t)LAYERS * N_EDGES * 4);
  float*    z      = (float*)   carve((size_t)E_TOT * 4);
  float*    exv    = (float*)   carve((size_t)E_TOT * 4);
  unsigned* mu     = (unsigned*)carve((size_t)N_NODES * 4);
  float*    sv     = (float*)   carve((size_t)N_NODES * 4);
  float*    p1     = (float*)   carve((size_t)N_NODES * 4);
  float*    p2     = (float*)   carve((size_t)N_NODES * 4);
  float*    pooled = (float*)   carve((size_t)NGRAPHS * HID * 4);
  float*    cnt    = (float*)   carve((size_t)NGRAPHS * 4);

  const int B = 256;
  auto nblk = [&](size_t n, int tpb) { return (unsigned)((n + tpb - 1) / tpb); };
  auto zero = [&](void* p, size_t n_u32) {
    fill_zero_kernel<<<nblk(n_u32, B), B, 0, stream>>>((uint32_t*)p, n_u32);
  };

  // one-time prep
  prep_wh_kernel<<<nblk((size_t)LAYERS * NPAD * KPAD, B), B, 0, stream>>>(Ws, Wh);
  prep_q_kernel<<<nblk((size_t)LAYERS * KPAD, B), B, 0, stream>>>(We, av, Q);
  edgeterm_kernel<<<nblk((size_t)N_EDGES, 8), B, 0, stream>>>(ea, Q, et);
  embed_kernel<<<nblk((size_t)N_NODES, 8), B, 0, stream>>>(x, emb, h, hh);

  const unsigned gemmBlocks = nblk((size_t)N_NODES / 16, 4); // 4 waves/block of 128
  for (int l = 0; l < LAYERS; ++l) {
    gemm_hx_kernel<<<gemmBlocks, 128, 0, stream>>>(hh, Wh + (size_t)l * NPAD * KPAD, hx, N_NODES);
    pscal_kernel<<<nblk((size_t)N_NODES, 8), B, 0, stream>>>(hx, av + (size_t)l * 2 * HID, p1, p2);
    zero(mu,   N_NODES);                 // 0 sorts below enc(-inf)
    zero(sv,   N_NODES);
    zero(outb, (size_t)N_NODES * HID);
    logits_kernel<<<nblk((size_t)E_TOT, B), B, 0, stream>>>(ei, et + (size_t)l * N_EDGES, p1, p2, z, mu);
    expsum_kernel<<<nblk((size_t)E_TOT, B), B, 0, stream>>>(ei, z, mu, exv, sv);
    scatter_kernel<<<nblk((size_t)E_TOT, 8), B, 0, stream>>>(ei, hx, exv, sv, outb);
    ln_kernel<<<nblk((size_t)N_NODES, 8), B, 0, stream>>>(outb, lng + (size_t)l * HID,
                                                          lnb + (size_t)l * HID, h, hh);
  }

  zero(pooled, (size_t)NGRAPHS * HID);
  zero(cnt,    (size_t)NGRAPHS);
  pool_kernel<<<nblk((size_t)N_NODES, 8), B, 0, stream>>>(bat, h, pooled, cnt);
  fc_kernel<<<nblk((size_t)NGRAPHS * OUTF, B), B, 0, stream>>>(pooled, cnt, fw, fb, outp);
}
